// InputAttention_45749991637067
// MI455X (gfx1250) — matrix operand order repaired and checked by reading, not compiled, not run on hardware
//
#include <hip/hip_runtime.h>
#include <math.h>

// T=512, N=2048, M=512, B=64
#define T_DIM 512
#define N_DIM 2048
#define B_DIM 64

typedef __attribute__((ext_vector_type(8)))  float  v8f;
typedef __attribute__((ext_vector_type(16))) __bf16 v16bf;

// Hardware tanh (CDNA5 V_TANH_F32 transcendental). ~1/4-rate trans op vs ~18-op libm
// polynomial: 67M evals -> ~6.5us instead of ~100us.
__device__ __forceinline__ float htanh(float x) {
#if __has_builtin(__builtin_amdgcn_tanhf)
  return __builtin_amdgcn_tanhf(x);
#elif __has_builtin(__builtin_amdgcn_tanh_f32)
  return __builtin_amdgcn_tanh_f32(x);
#else
  float r;
  // ISA 7.4.3: 1 independent op or V_NOP required after a TRANS op before its
  // output is used / sources overwritten. v_nop guarantees that here.
  asm volatile("v_tanh_f32 %0, %1\n\tv_nop" : "=v"(r) : "v"(x));
  return r;
#endif
}

// ---------------- q_proj = concat(h,s) @ W_e^T : [64,512] ----------------
__global__ __launch_bounds__(256)
void qproj_kernel(const float* __restrict__ h, const float* __restrict__ s,
                  const float* __restrict__ W, float* __restrict__ qp) {
  const int b = blockIdx.x;
  const int tid = threadIdx.x;
  __shared__ float hs[1024];
  for (int j = tid; j < 512; j += 256) {
    hs[j]       = h[b * 512 + j];
    hs[512 + j] = s[b * 512 + j];
  }
  __syncthreads();
  for (int u = tid; u < 512; u += 256) {
    const float4* w4 = (const float4*)(W + (size_t)u * 1024);
    float acc = 0.f;
    for (int j = 0; j < 256; ++j) {
      float4 wv = w4[j];
      acc += wv.x * hs[4*j] + wv.y * hs[4*j+1] + wv.z * hs[4*j+2] + wv.w * hs[4*j+3];
    }
    qp[b * 512 + u] = acc;
  }
}

// ---------------- U_e (f32, [512][512] row-major) -> bf16, same layout ----------------
__global__ __launch_bounds__(256)
void cvt_u_kernel(const float* __restrict__ U, unsigned short* __restrict__ ub_raw) {
  __bf16* ub = (__bf16*)ub_raw;
  int i = blockIdx.x * 256 + threadIdx.x;
  for (; i < 512 * 512; i += gridDim.x * 256) ub[i] = (__bf16)U[i];
}

// ---------------- main: per (b, 128-row n-block) GEMM + tanh + dot(v) ----------------
// Zero LDS: each wave owns 16 n-rows; A (16x512 bf16 = 128 VGPRs/lane) is built
// directly from global with immediate-offset b32 loads and reused across all 32 u-tiles.
extern "C" __global__ __launch_bounds__(256)
void attn_main_kernel(const float* __restrict__ data,
                      const unsigned short* __restrict__ ubf_raw,
                      const float* __restrict__ qp,
                      const float* __restrict__ ve,
                      float* __restrict__ scores) {
  const __bf16* ubf = (const __bf16*)ubf_raw;

  const int b    = blockIdx.y;
  const int n0   = blockIdx.x * 128;
  const int tid  = threadIdx.x;
  const int lane = tid & 31;
  const int wave = tid >> 5;       // 8 waves; wave owns rows [wave*16, wave*16+16)
  const int lm   = lane & 15;      // A: row M; B/C: column N (=u)
  const int lh   = lane >> 4;      // lane half -> K sub-block select

  // Per-lane A base: fixed column n_row of data[b] (t-stride = N_DIM floats = 8KB,
  // all 256 element offsets fit the 24-bit immediate field).
  const int n_row = n0 + wave * 16 + lm;
  const float* abase = data + (size_t)b * T_DIM * N_DIM + n_row
                     + (size_t)(lh * 16) * N_DIM;  // fold lane-half K offset into base

  // Build 16 A-fragments: fragment ks covers K = ks*32..ks*32+31.
  // Lane halfword h -> K = ks*32 + 16*lh + h
  v16bf afrag[16];
  #pragma unroll
  for (int ks = 0; ks < 16; ++ks) {
    v16bf A;
    #pragma unroll
    for (int hh = 0; hh < 8; ++hh) {
      A[hh]     = (__bf16)abase[(size_t)(ks * 32 + hh) * N_DIM];
      A[hh + 8] = (__bf16)abase[(size_t)(ks * 32 + 8 + hh) * N_DIM];
    }
    afrag[ks] = A;
  }

  float partial[8];
  #pragma unroll
  for (int r = 0; r < 8; ++r) partial[r] = 0.f;

  for (int ut = 0; ut < 32; ++ut) {
    const int u    = ut * 16 + lm;               // this lane's u column
    const float qv = qp[b * 512 + u];
    const float vv = ve[u];
    const __bf16* brow = ubf + (size_t)u * 512 + lh * 16;  // B: K = ks*32 + 16*lh + h

    v8f acc = {};
    #pragma unroll
    for (int ks = 0; ks < 16; ++ks) {
      v16bf Bm = *(const v16bf*)(brow + ks * 32);
      acc = __builtin_amdgcn_wmma_f32_16x16x32_bf16(
          /*neg_a=*/false, afrag[ks], /*neg_b=*/false, Bm,
          /*c_mod=*/(short)0, acc, /*reuse_a=*/false, /*reuse_b=*/false);
    }
    // C layout: VGPR r = row r (lanes 0-15) / row r+8 (lanes 16-31), col = u0+lm.
    #pragma unroll
    for (int r = 0; r < 8; ++r)
      partial[r] += htanh(qv + acc[r]) * vv;
  }

  // Reduce over the 16 u-columns held within each lane half (halves stay separate
  // because xor masks < 16).
  #pragma unroll
  for (int r = 0; r < 8; ++r) {
    float x = partial[r];
    x += __shfl_xor(x, 1, 32);
    x += __shfl_xor(x, 2, 32);
    x += __shfl_xor(x, 4, 32);
    x += __shfl_xor(x, 8, 32);
    partial[r] = x;
  }
  if (lm == 0) {  // lane 0 -> rows 0..7, lane 16 -> rows 8..15
    const int gn = n0 + wave * 16 + lh * 8;
    #pragma unroll
    for (int r = 0; r < 8; ++r)
      scores[(size_t)b * N_DIM + gn + r] = partial[r];
  }
}

// ---------------- softmax over n (2048) per batch ----------------
__global__ __launch_bounds__(256)
void softmax_kernel(const float* __restrict__ scores, float* __restrict__ out) {
  const int b = blockIdx.x;
  const int tid = threadIdx.x;
  const int lane = tid & 31, wave = tid >> 5;
  __shared__ float rmax[8], rsum[8];
  const float* s = scores + (size_t)b * N_DIM;

  float vals[8];
  float m = -3.4e38f;
  #pragma unroll
  for (int k = 0; k < 8; ++k) { vals[k] = s[tid + k * 256]; m = fmaxf(m, vals[k]); }
  #pragma unroll
  for (int d = 16; d >= 1; d >>= 1) m = fmaxf(m, __shfl_xor(m, d, 32));
  if (lane == 0) rmax[wave] = m;
  __syncthreads();
  m = rmax[0];
  #pragma unroll
  for (int w = 1; w < 8; ++w) m = fmaxf(m, rmax[w]);

  float sum = 0.f;
  #pragma unroll
  for (int k = 0; k < 8; ++k) { vals[k] = expf(vals[k] - m); sum += vals[k]; }
  #pragma unroll
  for (int d = 16; d >= 1; d >>= 1) sum += __shfl_xor(sum, d, 32);
  if (lane == 0) rsum[wave] = sum;
  __syncthreads();
  sum = 0.f;
  #pragma unroll
  for (int w = 0; w < 8; ++w) sum += rsum[w];

  const float inv = 1.f / sum;
  float* o = out + (size_t)b * N_DIM;
  #pragma unroll
  for (int k = 0; k < 8; ++k) o[tid + k * 256] = vals[k] * inv;
}

// ---------------- launch ----------------
extern "C" void kernel_launch(void* const* d_in, const int* in_sizes, int n_in,
                              void* d_out, int out_size, void* d_ws, size_t ws_size,
                              hipStream_t stream) {
  const float* h    = (const float*)d_in[0];  // h_t  [1,64,512]
  const float* s    = (const float*)d_in[1];  // s_t  [1,64,512]
  const float* data = (const float*)d_in[2];  // data [64,512,2048]
  const float* We   = (const float*)d_in[3];  // W_e  [512,1024]
  const float* Ue   = (const float*)d_in[4];  // U_e  [512,512]
  const float* ve   = (const float*)d_in[5];  // v_e  [1,512]

  // workspace: qp (128KB) | U_bf16 (512KB) | scores (512KB)  => ~1.13 MB total
  char* ws = (char*)d_ws;
  float*          qp     = (float*)(ws + 0);
  unsigned short* ubf    = (unsigned short*)(ws + 131072);
  float*          scores = (float*)(ws + 131072 + 524288);
  float*          out    = (float*)d_out;

  qproj_kernel<<<dim3(B_DIM), dim3(256), 0, stream>>>(h, s, We, qp);
  cvt_u_kernel<<<dim3(256), dim3(256), 0, stream>>>(Ue, ubf);
  attn_main_kernel<<<dim3(N_DIM / 128, B_DIM), dim3(256), 0, stream>>>(
      data, ubf, qp, ve, scores);
  softmax_kernel<<<dim3(B_DIM), dim3(256), 0, stream>>>(scores, out);
}
